// SSMLayer_49374944034993
// MI455X (gfx1250) — matrix operand order repaired and checked by reading, not compiled, not run on hardware
//
#include <hip/hip_runtime.h>
#include <hip/hip_bf16.h>

#define SEQ 2048
#define D_MODEL 1024
#define D_INNER 2048
#define D_STATE 16
#define D_CONV 4

typedef __bf16 bf16;
typedef __attribute__((ext_vector_type(16))) __bf16 v16bf;
typedef __attribute__((ext_vector_type(8)))  __bf16 v8bf;
typedef __attribute__((ext_vector_type(8)))  float  v8f;

typedef unsigned int u32;
typedef unsigned long long u64;
typedef __attribute__((ext_vector_type(4))) u32 v4u32;
typedef __attribute__((ext_vector_type(8))) int v8i32;
typedef __attribute__((ext_vector_type(4))) int v4i32;

#if defined(__HIP_DEVICE_COMPILE__) && __has_builtin(__builtin_amdgcn_tensor_load_to_lds)
#define HAVE_TDM 1
#else
#define HAVE_TDM 0
#endif

// ---------------------------------------------------------------------------
// WMMA fragment loaders (wave32, 16x16x32 bf16)
// A: 16x32 (MxK) row-major, lda=K. lane L: row=L&15, kg=L>>4 holds
//   K = kg*8..+7 (VGPR0..3) and K = 16+kg*8..+7 (VGPR4..7): two 16B loads.
// B: 32x16 (KxN) from pre-transposed Wt[N,K]: lane L: col n=L&15, kg=L>>4
//   holds K = kg*8..+7 and K = 16+kg*8..+7 (same as A).
// ---------------------------------------------------------------------------
__device__ __forceinline__ v16bf load_frag_a(const bf16* __restrict__ p, int ld) {
  const int lane = threadIdx.x & 31;
  const bf16* q = p + (size_t)(lane & 15) * ld + (lane >> 4) * 8;
  v16bf f;
  ((v8bf*)&f)[0] = *(const v8bf*)(q);
  ((v8bf*)&f)[1] = *(const v8bf*)(q + 16);
  return f;
}

__device__ __forceinline__ v16bf load_frag_b(const bf16* __restrict__ p, int ld) {
  const int lane = threadIdx.x & 31;
  const bf16* q = p + (size_t)(lane & 15) * ld + (lane >> 4) * 8;
  v16bf f;
  ((v8bf*)&f)[0] = *(const v8bf*)(q);
  ((v8bf*)&f)[1] = *(const v8bf*)(q + 16);
  return f;
}

// ---------------------------------------------------------------------------
// GEMM: C[M,N] = A[M,K](bf16) * Bt[N,K](bf16)^T (+ add)
// Block = 256 threads = 8 waves arranged 2(M) x 4(N); wave tile = 64x64
// (16 v_wmma per K-step, 16 b128 loads -> 1.0 loads/wmma).
// Grid: (N/256, M/128).
// ---------------------------------------------------------------------------
__global__ __launch_bounds__(256) void gemm_bf16_wmma(
    const bf16* __restrict__ A, const bf16* __restrict__ Bt,
    float* __restrict__ C, const float* __restrict__ add,
    int M, int N, int K)
{
  const int lane = threadIdx.x & 31;
  const int wave = threadIdx.x >> 5;
  const int m0 = blockIdx.y * 128 + (wave >> 2) * 64;
  const int n0 = blockIdx.x * 256 + (wave & 3) * 64;

  const v8f vzero = {0.f,0.f,0.f,0.f,0.f,0.f,0.f,0.f};
  v8f acc[4][4];
#pragma unroll
  for (int i = 0; i < 4; ++i)
#pragma unroll
    for (int j = 0; j < 4; ++j) acc[i][j] = vzero;

  for (int k0 = 0; k0 < K; k0 += 32) {
    if (k0 + 32 < K) {
      __builtin_prefetch(A  + (size_t)m0 * K + k0 + 32, 0, 1);   // global_prefetch_b8
      __builtin_prefetch(Bt + (size_t)n0 * K + k0 + 32, 0, 1);
    }
    v16bf a[4], b[4];
#pragma unroll
    for (int i = 0; i < 4; ++i)
      a[i] = load_frag_a(A + (size_t)(m0 + 16 * i) * K + k0, K);
#pragma unroll
    for (int j = 0; j < 4; ++j)
      b[j] = load_frag_b(Bt + (size_t)(n0 + 16 * j) * K + k0, K);
#pragma unroll
    for (int i = 0; i < 4; ++i)
#pragma unroll
      for (int j = 0; j < 4; ++j)
        acc[i][j] = __builtin_amdgcn_wmma_f32_16x16x32_bf16(
            false, a[i], false, b[j], (short)0, acc[i][j], false, false);
  }

  // C/D layout: VGPR r -> M = r + (lane>=16 ? 8 : 0), N = lane&15
  const int n  = lane & 15;
  const int mb = (lane >> 4) * 8;
#pragma unroll
  for (int i = 0; i < 4; ++i)
#pragma unroll
    for (int j = 0; j < 4; ++j)
#pragma unroll
      for (int r = 0; r < 8; ++r) {
        const int row = m0 + i * 16 + mb + r;
        const int col = n0 + j * 16 + n;
        float v = acc[i][j][r];
        if (add) v += add[(size_t)row * N + col];
        C[(size_t)row * N + col] = v;
      }
}

// ---------------------------------------------------------------------------
// RMSNorm over D_MODEL, emit bf16. One block (256 thr) per token.
// ---------------------------------------------------------------------------
__global__ __launch_bounds__(256) void rmsnorm_bf16(
    const float* __restrict__ x, const float* __restrict__ w,
    bf16* __restrict__ out)
{
  const int t   = blockIdx.x;
  const int tid = threadIdx.x;
  const float* xr = x + (size_t)t * D_MODEL;
  float v[4];
  float s = 0.f;
#pragma unroll
  for (int i = 0; i < 4; ++i) { v[i] = xr[tid + i * 256]; s += v[i] * v[i]; }
  __shared__ float red[256];
  red[tid] = s;
  __syncthreads();
  for (int st = 128; st > 0; st >>= 1) {
    if (tid < st) red[tid] += red[tid + st];
    __syncthreads();
  }
  const float rs = rsqrtf(red[0] / (float)D_MODEL + 1e-6f);
#pragma unroll
  for (int i = 0; i < 4; ++i) {
    const int c = tid + i * 256;
    out[(size_t)t * D_MODEL + c] = (bf16)(v[i] * rs * w[c]);
  }
}

// ---------------------------------------------------------------------------
// W[K,N] fp32 row-major -> Wt[N,K] bf16 row-major
// ---------------------------------------------------------------------------
__global__ void convert_transpose(const float* __restrict__ W,
                                  bf16* __restrict__ Wt, int K, int N)
{
  const size_t idx = (size_t)blockIdx.x * blockDim.x + threadIdx.x;
  if (idx >= (size_t)K * N) return;
  const int k = (int)(idx / N);
  const int n = (int)(idx % N);
  Wt[(size_t)n * K + k] = (bf16)W[idx];
}

// ---------------------------------------------------------------------------
// Causal depthwise conv1d (k=4) + bias + SiLU. xb = xz[:, 0:D_INNER].
// ---------------------------------------------------------------------------
__global__ void conv_silu(const float* __restrict__ xz,
                          const float* __restrict__ cw,
                          const float* __restrict__ cb,
                          float* __restrict__ xc)
{
  const size_t idx = (size_t)blockIdx.x * blockDim.x + threadIdx.x;
  if (idx >= (size_t)SEQ * D_INNER) return;
  const int s = (int)(idx >> 11);          // D_INNER == 2048
  const int c = (int)(idx & (D_INNER - 1));
  float acc = cb[c];
#pragma unroll
  for (int j = 0; j < D_CONV; ++j) {
    const int sp = s - (D_CONV - 1) + j;
    if (sp >= 0) acc += cw[c * D_CONV + j] * xz[(size_t)sp * (2 * D_INNER) + c];
  }
  xc[idx] = acc / (1.f + __expf(-acc));     // SiLU
}

// ---------------------------------------------------------------------------
// x_proj: x_dbl[s,o] = sum_k xc[s,k] * w[k,o]   (N=33 skinny; wave per output)
// ---------------------------------------------------------------------------
__global__ __launch_bounds__(256) void xproj(const float* __restrict__ xc,
                                             const float* __restrict__ w,
                                             float* __restrict__ xdbl)
{
  const int NOUT = 2 * D_STATE + 1;  // 33
  const int gw   = (int)(((size_t)blockIdx.x * blockDim.x + threadIdx.x) >> 5);
  const int lane = threadIdx.x & 31;
  if (gw >= SEQ * NOUT) return;      // wave-uniform exit
  const int s = gw / NOUT;
  const int o = gw % NOUT;
  const float* xr = xc + (size_t)s * D_INNER;
  float acc = 0.f;
  for (int k = lane; k < D_INNER; k += 32) acc += xr[k] * w[k * NOUT + o];
  for (int off = 16; off > 0; off >>= 1) acc += __shfl_down(acc, off, 32);
  if (lane == 0) xdbl[(size_t)s * NOUT + o] = acc;
}

// ---------------------------------------------------------------------------
// Tensor Data Mover: 1D fp32 copy global -> LDS (contiguous slab).
// D# packing per CDNA5 ISA ch.8 (group0: count/lds/global/type, group1:
// data_size=4B, tensor_dim0=tile_dim0=nelem, dim1=1, stride0=nelem).
// This toolchain exposes the 6-arg builtin:
//   (uint32x4 g0, int32x8 g1, int32x4 g2, int32x4 g3, int32x8 g4, i32 cpol)
// Groups 2-4 are only needed for >2D tensors; zero-filled here.
// ---------------------------------------------------------------------------
#if HAVE_TDM
__device__ __forceinline__ void tdm_load_f32_1d(const void* lds_ptr,
                                                const float* gptr, u32 nelem)
{
  const u64 ga = (u64)(uintptr_t)gptr;
  const u32 lo = (u32)(uintptr_t)lds_ptr;      // generic LDS addr: low 32 = offset
  v4u32 g0;
  g0[0] = 1u;                                   // count=1, no gather
  g0[1] = lo;                                   // lds_addr (bytes)
  g0[2] = (u32)ga;                              // global_addr[31:0]
  g0[3] = (u32)((ga >> 32) & 0x01FFFFFFu) | (2u << 30);  // addr[56:32] | type=2
  v8i32 g1;
  g1[0] = (int)(2u << 16);                      // wg_mask=0 | data_size=2 (4B)
  g1[1] = (int)((nelem & 0xFFFFu) << 16);       // tensor_dim0[15:0]  @ [31:16]
  g1[2] = (int)((nelem >> 16) & 0xFFFFu) | (int)(1u << 16); // dim0[31:16] | tensor_dim1=1
  g1[3] = (int)((nelem & 0xFFFFu) << 16);       // tensor_dim1 hi=0 | tile_dim0
  g1[4] = 1;                                    // tile_dim1=1, tile_dim2=0
  g1[5] = (int)nelem;                           // tensor_dim0_stride[31:0]
  g1[6] = 0;                                    // stride0 hi | stride1 lo
  g1[7] = 0;                                    // stride1 hi
  const v4i32 z4 = {0, 0, 0, 0};
  const v8i32 z8 = {0, 0, 0, 0, 0, 0, 0, 0};
  __builtin_amdgcn_tensor_load_to_lds(g0, g1, z4, z4, z8, 0);
}
#endif

// ---------------------------------------------------------------------------
// Sequential selective scan fused with D-skip + SiLU(z) gating -> bf16.
// One thread per channel, h[16] in registers. x_dbl rows are staged in LDS
// in 128-row slabs, double-buffered via the Tensor Data Mover (TENSORcnt),
// so the 128-step inner loop has zero barriers.
// ---------------------------------------------------------------------------
#define SCAN_CH 128
#define SCAN_NC (SEQ / SCAN_CH)

__global__ __launch_bounds__(256) void ssm_scan(
    const float* __restrict__ xdbl, const float* __restrict__ xc,
    const float* __restrict__ xz,   const float* __restrict__ A_log,
    const float* __restrict__ Dp,   const float* __restrict__ dtw,
    const float* __restrict__ dtb,  bf16* __restrict__ ybf)
{
  const int c = blockIdx.x * 256 + threadIdx.x;
  float Ac[D_STATE];
#pragma unroll
  for (int j = 0; j < D_STATE; ++j) Ac[j] = -__expf(A_log[(size_t)c * D_STATE + j]);
  const float wdt = dtw[c], bdt = dtb[c], Dc = Dp[c];
  float h[D_STATE];
#pragma unroll
  for (int j = 0; j < D_STATE; ++j) h[j] = 0.f;

  __shared__ float slab[2][SCAN_CH * 33];

#if HAVE_TDM
  if (threadIdx.x < 32)             // single wave issues TDM (branch-skipped elsewhere)
    tdm_load_f32_1d(&slab[0][0], xdbl, SCAN_CH * 33);
#endif

  for (int ck = 0; ck < SCAN_NC; ++ck) {
#if HAVE_TDM
    if (threadIdx.x < 32)
      __builtin_amdgcn_s_wait_tensorcnt(0);   // slab ck landed in LDS
    __syncthreads();                          // visible to all waves; prev buffer free
    if (threadIdx.x < 32 && ck + 1 < SCAN_NC)
      tdm_load_f32_1d(&slab[(ck + 1) & 1][0],
                      xdbl + (size_t)(ck + 1) * SCAN_CH * 33, SCAN_CH * 33);
    const float* buf = slab[ck & 1];
#else
    __syncthreads();                          // all waves done with previous slab
    for (int i = threadIdx.x; i < SCAN_CH * 33; i += 256)
      slab[0][i] = xdbl[(size_t)ck * SCAN_CH * 33 + i];
    __syncthreads();
    const float* buf = slab[0];
#endif

    for (int t = 0; t < SCAN_CH; ++t) {
      const int s = ck * SCAN_CH + t;
      const float* row = buf + t * 33;
      const float v  = row[0] * wdt + bdt;
      const float dt = (v > 20.f) ? v : log1pf(__expf(v));   // softplus
      const float xv = xc[(size_t)s * D_INNER + c];
      float y = 0.f;
#pragma unroll
      for (int j = 0; j < D_STATE; ++j) {
        const float dA = __expf(dt * Ac[j]);
        h[j] = dA * h[j] + dt * row[1 + j] * xv;
        y += h[j] * row[17 + j];
      }
      const float zv   = xz[(size_t)s * (2 * D_INNER) + D_INNER + c];
      const float gate = zv / (1.f + __expf(-zv));
      ybf[(size_t)s * D_INNER + c] = (bf16)((y + xv * Dc) * gate);
    }
  }
}

// ---------------------------------------------------------------------------
extern "C" void kernel_launch(void* const* d_in, const int* in_sizes, int n_in,
                              void* d_out, int out_size, void* d_ws, size_t ws_size,
                              hipStream_t stream) {
  (void)in_sizes; (void)n_in; (void)out_size; (void)ws_size;
  const float* x         = (const float*)d_in[0];
  const float* norm_w    = (const float*)d_in[1];
  const float* in_proj_w = (const float*)d_in[2];
  const float* conv_w    = (const float*)d_in[3];
  const float* conv_b    = (const float*)d_in[4];
  const float* A_log     = (const float*)d_in[5];
  const float* Dp        = (const float*)d_in[6];
  const float* x_proj_w  = (const float*)d_in[7];
  const float* dt_proj_w = (const float*)d_in[8];
  const float* dt_proj_b = (const float*)d_in[9];
  const float* out_proj_w= (const float*)d_in[10];
  float* out = (float*)d_out;

  char* ws = (char*)d_ws;
  size_t off = 0;
  auto alloc = [&](size_t bytes) -> void* {
    void* p = ws + off;
    off = (off + bytes + 255) & ~(size_t)255;
    return p;
  };
  bf16*  xn    = (bf16*) alloc((size_t)SEQ * D_MODEL * 2);
  bf16*  inWt  = (bf16*) alloc((size_t)2 * D_INNER * D_MODEL * 2);
  bf16*  outWt = (bf16*) alloc((size_t)D_MODEL * D_INNER * 2);
  float* xz    = (float*)alloc((size_t)SEQ * 2 * D_INNER * 4);
  float* xc    = (float*)alloc((size_t)SEQ * D_INNER * 4);
  float* xdbl  = (float*)alloc((size_t)SEQ * 33 * 4);
  bf16*  ybf   = (bf16*) alloc((size_t)SEQ * D_INNER * 2);

  // 1) RMSNorm -> bf16
  rmsnorm_bf16<<<SEQ, 256, 0, stream>>>(x, norm_w, xn);

  // 2) Weight convert+transpose to bf16 [N,K]
  {
    const size_t n1 = (size_t)D_MODEL * 2 * D_INNER;
    convert_transpose<<<(unsigned)((n1 + 255) / 256), 256, 0, stream>>>(
        in_proj_w, inWt, D_MODEL, 2 * D_INNER);
    const size_t n2 = (size_t)D_INNER * D_MODEL;
    convert_transpose<<<(unsigned)((n2 + 255) / 256), 256, 0, stream>>>(
        out_proj_w, outWt, D_INNER, D_MODEL);
  }

  // 3) in_proj GEMM: [SEQ,D_MODEL] x [D_MODEL,2*D_INNER] -> xz fp32
  gemm_bf16_wmma<<<dim3((2 * D_INNER) / 256, SEQ / 128), 256, 0, stream>>>(
      xn, inWt, xz, nullptr, SEQ, 2 * D_INNER, D_MODEL);

  // 4) causal depthwise conv + SiLU
  {
    const size_t n = (size_t)SEQ * D_INNER;
    conv_silu<<<(unsigned)((n + 255) / 256), 256, 0, stream>>>(xz, conv_w, conv_b, xc);
  }

  // 5) x_proj skinny GEMM (wave per output)
  {
    const size_t waves = (size_t)SEQ * 33;
    xproj<<<(unsigned)((waves * 32 + 255) / 256), 256, 0, stream>>>(xc, x_proj_w, xdbl);
  }

  // 6) selective scan (TDM-staged x_dbl slabs, + D skip + SiLU(z) gate) -> bf16
  ssm_scan<<<D_INNER / 256, 256, 0, stream>>>(xdbl, xc, xz, A_log, Dp,
                                              dt_proj_w, dt_proj_b, ybf);

  // 7) out_proj GEMM + residual -> d_out fp32
  gemm_bf16_wmma<<<dim3(D_MODEL / 256, SEQ / 128), 256, 0, stream>>>(
      ybf, outWt, out, x, SEQ, D_MODEL, D_INNER);
}
